// LlamaAttention_23656679867621
// MI455X (gfx1250) — matrix operand order repaired and checked by reading, not compiled, not run on hardware
//
#include <hip/hip_runtime.h>

// Problem constants (B=1)
#define L_SEQ 2048
#define HIDN  1024
#define NH    16
#define DHD   64
#define QS    (NH * DHD)         // 1024
#define EPSF  1.1920928955078125e-07f   // float32 machine eps

typedef __attribute__((ext_vector_type(16))) _Float16 v16h;
typedef __attribute__((ext_vector_type(8)))  float    v8f;
typedef __attribute__((ext_vector_type(4)))  unsigned u32x4;
typedef __attribute__((ext_vector_type(8)))  int      i32x8;
typedef __attribute__((ext_vector_type(4)))  int      i32x4;

// ---------------------------------------------------------------------------
// WMMA fragment loaders (wave32, v_wmma_f32_16x16x32_f16 operand layouts)
// A: 16x32 (MxK) f16. lane 0-15 -> row=lane, K chunks {0..7,16..23};
//                     lane 16-31 -> row=lane-16, K chunks {8..15,24..31}.
// B: 32x16 (KxN) f16 given as [N][K] in LDS (i.e. B^T row-major).
//    lane 0-15 -> col=lane, K {0..7,16..23}; lane 16-31 -> col=lane-16, K {8..15,24..31}.
// C/D: VGPR r -> (M = r + (lane<16?0:8), N = lane&15).
// ---------------------------------------------------------------------------
__device__ __forceinline__ v16h frag_a_lds(const _Float16* p, int ld) {
  int lane = threadIdx.x & 31;
  const _Float16* r = p + (lane & 15) * ld;
  int kb = (lane & 16) ? 8 : 0;
  v16h f;
#pragma unroll
  for (int h = 0; h < 8; ++h) f[h] = r[kb + h];
#pragma unroll
  for (int h = 0; h < 8; ++h) f[8 + h] = r[16 + kb + h];
  return f;
}

__device__ __forceinline__ v16h frag_b_lds(const _Float16* p, int ld) {
  int lane = threadIdx.x & 31;
  const _Float16* c = p + (lane & 15) * ld + ((lane & 16) ? 8 : 0);
  v16h f;
#pragma unroll
  for (int h = 0; h < 8; ++h) { f[h] = c[h]; f[8 + h] = c[16 + h]; }
  return f;
}

// One 16-byte async global->LDS copy (ASYNCcnt-tracked). Generic pointers to
// __shared__ carry the LDS byte offset in their low 32 bits.
__device__ __forceinline__ void async_copy_b128(const void* gptr, void* lptr) {
  unsigned lds = (unsigned)(unsigned long long)lptr;
  unsigned long long ga = (unsigned long long)gptr;
  asm volatile("global_load_async_to_lds_b128 %0, %1, off"
               :: "v"(lds), "v"(ga) : "memory");
}
__device__ __forceinline__ void wait_async0() {
  asm volatile("s_wait_asynccnt 0x0" ::: "memory");
}

// ---------------------------------------------------------------------------
// Tensor Data Mover: DMA one 2-D tile of 32 rows x 64 f16 (row stride 64)
// from global into LDS.  D# per CDNA5 ISA Sec 8.3/8.4:
//  group0: count=1 | lds_addr[63:32] | global_addr[120:64] | type=2
//  group1: data_size=2B, tensor_dim0=64, tensor_dim1=32, tile_dim0=64,
//          tile_dim1=32, tensor_dim0_stride=64
// ---------------------------------------------------------------------------
__device__ __forceinline__ void tdm_load_tile_32x64_f16(unsigned lds_addr,
                                                        unsigned long long ga) {
  u32x4 g0;
  g0[0] = 1u;                                           // count=1
  g0[1] = lds_addr;                                     // lds_addr
  g0[2] = (unsigned)ga;                                 // global_addr lo
  g0[3] = (unsigned)((ga >> 32) & 0x01FFFFFFull) | 0x80000000u;  // hi|type=2
  i32x8 g1;
  g1[0] = 0x00010000;          // workgroup_mask=0, data_size=1 (2 bytes)
  g1[1] = (int)(64u << 16);    // tensor_dim0 = 64   (bits 63:48)
  g1[2] = (int)(32u << 16);    // tensor_dim1 = 32   (bits 95:80)
  g1[3] = (int)(64u << 16);    // tile_dim0   = 64   (bits 127:112)
  g1[4] = 32;                  // tile_dim1   = 32   (bits 143:128)
  g1[5] = 64;                  // tensor_dim0_stride = 64 (bits 191:160)
  g1[6] = 0;
  g1[7] = 0;
  i32x4 gz = {0, 0, 0, 0};
#if __has_include(<hip/amd_detail/amd_gfx1250_TDM.h>)
  i32x8 gz8 = {0, 0, 0, 0, 0, 0, 0, 0};
  __builtin_amdgcn_tensor_load_to_lds(g0, g1, gz, gz, gz8, 0);
#else
  __builtin_amdgcn_tensor_load_to_lds(g0, g1, gz, gz, 0);
#endif
}

// ---------------------------------------------------------------------------
// Generic GEMM: C[M][NN] = A[M][KK] * B[NN][KK]^T (f32 in/out, f16 WMMA math)
// 128x128 tile per 256-thread block (8 waves, each 32x64), K-step 32.
// ---------------------------------------------------------------------------
template <int NN, int KK>
__global__ __launch_bounds__(256) void gemm_f16wmma(const float* __restrict__ A,
                                                    const float* __restrict__ Bm,
                                                    float* __restrict__ C) {
  __shared__ _Float16 As[128][40];
  __shared__ _Float16 Bs[128][40];
  const int tid  = threadIdx.x;
  const int wave = tid >> 5, lane = tid & 31;
  const int bm = blockIdx.y * 128, bn = blockIdx.x * 128;
  const int wm = (wave >> 1) * 32, wn = (wave & 1) * 64;

  v8f vz = {};
  v8f acc[2][4];
#pragma unroll
  for (int i = 0; i < 2; ++i)
#pragma unroll
    for (int j = 0; j < 4; ++j) acc[i][j] = vz;

  const int lrow = tid >> 1;
  const int lcb  = (tid & 1) * 16;
  const float* ag = A  + (size_t)(bm + lrow) * KK + lcb;
  const float* bg = Bm + (size_t)(bn + lrow) * KK + lcb;

  float ar[16], br[16];
#pragma unroll
  for (int e = 0; e < 16; ++e) { ar[e] = ag[e]; br[e] = bg[e]; }

  for (int ks = 0; ks < KK; ks += 32) {
#pragma unroll
    for (int e = 0; e < 16; ++e) {
      As[lrow][lcb + e] = (_Float16)ar[e];
      Bs[lrow][lcb + e] = (_Float16)br[e];
    }
    __syncthreads();
    if (ks + 32 < KK) {          // register-prefetch next K-step (uniform)
#pragma unroll
      for (int e = 0; e < 16; ++e) {
        ar[e] = ag[ks + 32 + e];
        br[e] = bg[ks + 32 + e];
      }
    }
#pragma unroll
    for (int mt = 0; mt < 2; ++mt) {
      v16h af = frag_a_lds(&As[wm + mt * 16][0], 40);
#pragma unroll
      for (int nt = 0; nt < 4; ++nt) {
        v16h bf = frag_b_lds(&Bs[wn + nt * 16][0], 40);
        acc[mt][nt] = __builtin_amdgcn_wmma_f32_16x16x32_f16(
            false, af, false, bf, (short)0, acc[mt][nt], false, false);
      }
    }
    __syncthreads();
  }

  const int rbase = (lane & 16) ? 8 : 0;
  const int cofs  = lane & 15;
#pragma unroll
  for (int mt = 0; mt < 2; ++mt)
#pragma unroll
    for (int nt = 0; nt < 4; ++nt) {
      float* cp = C + (size_t)(bm + wm + mt * 16 + rbase) * NN
                    + (bn + wn + nt * 16 + cofs);
#pragma unroll
      for (int r = 0; r < 8; ++r) cp[r * NN] = acc[mt][nt][r];
    }
}

// ---------------------------------------------------------------------------
// dt = softplus(hidden @ dt_proj_w^T + dt_proj_b + dt_bias), (L, NH)
// ---------------------------------------------------------------------------
__global__ __launch_bounds__(256) void dt_kernel(const float* __restrict__ hidden,
                                                 const float* __restrict__ dtw,
                                                 const float* __restrict__ dtb,
                                                 const float* __restrict__ dt_bias,
                                                 float* __restrict__ dt_out) {
  int t = blockIdx.x * 256 + threadIdx.x;   // 0 .. L*NH-1
  int l = t / NH, h = t % NH;
  const float* hr = hidden + (size_t)l * HIDN;
  const float* wr = dtw + (size_t)h * HIDN;
  float s = 0.f;
  for (int k = 0; k < HIDN; ++k) s += hr[k] * wr[k];
  s += dtb[h] + dt_bias[h];
  float dt = (s > 20.f) ? s : log1pf(__expf(s));
  dt_out[t] = dt;
}

// ---------------------------------------------------------------------------
// acum[h][l] = cumsum_l( -exp(A_log[h]) * dt[l][h] )
// ---------------------------------------------------------------------------
__global__ void cumsum_kernel(const float* __restrict__ dt,
                              const float* __restrict__ A_log,
                              float* __restrict__ acum) {
  int h = threadIdx.x;                      // <<<1, 16>>>
  float nA = -__expf(A_log[h]);
  float acc = 0.f;
  for (int l = 0; l < L_SEQ; ++l) {
    acc += nA * dt[l * NH + h];
    acum[h * L_SEQ + l] = acc;
  }
}

// ---------------------------------------------------------------------------
// Causal depthwise conv (K=2) + head split + dt-scaling of V; f32 -> f16.
// Outputs q/k/v as [NH][L][DH] f16.
// ---------------------------------------------------------------------------
__global__ __launch_bounds__(256) void conv_split_kernel(
    const float* __restrict__ qkv,
    const float* __restrict__ qw, const float* __restrict__ qb,
    const float* __restrict__ kw, const float* __restrict__ kb_,
    const float* __restrict__ vw, const float* __restrict__ vb,
    const float* __restrict__ dt,
    _Float16* __restrict__ qf, _Float16* __restrict__ kf,
    _Float16* __restrict__ vf) {
  int idx = blockIdx.x * 256 + threadIdx.x;   // L*QS total
  int l = idx / QS, c = idx % QS;
  int h = c / DHD, d = c % DHD;
  const float* row = qkv + (size_t)l * (3 * QS);
  float xq1 = row[c], xk1 = row[QS + c], xv1 = row[2 * QS + c];
  float xq0 = 0.f, xk0 = 0.f, xv0 = 0.f;
  if (l > 0) {
    const float* pr = row - 3 * QS;
    xq0 = pr[c]; xk0 = pr[QS + c]; xv0 = pr[2 * QS + c];
  }
  float qv = qb[c]  + xq0 * qw[2 * c] + xq1 * qw[2 * c + 1];
  float kv = kb_[c] + xk0 * kw[2 * c] + xk1 * kw[2 * c + 1];
  float vv = (vb[c] + xv0 * vw[2 * c] + xv1 * vw[2 * c + 1]) * dt[l * NH + h];
  size_t o = ((size_t)h * L_SEQ + l) * DHD + d;
  qf[o] = (_Float16)qv;
  kf[o] = (_Float16)kv;
  vf[o] = (_Float16)vv;
}

// ---------------------------------------------------------------------------
// Flash-style gated attention + RMSNorm.
// Grid (16 row-tiles, 16 heads), 256 threads (8 waves), wave w owns rows
// w*16..w*16+15 and the full DH=64 output.  Per 32-key tile:
//   - K tile DMA'd by the Tensor Data Mover (wave 0, TENSORcnt)
//   - V tile async-copied raw (ASYNCcnt), transposed LDS->LDS
//   - S = Q K^T (WMMA), exp-gate on f32 accum, O += P V (WMMA)
// ---------------------------------------------------------------------------
__global__ __launch_bounds__(256) void attn_kernel(
    const _Float16* __restrict__ qf, const _Float16* __restrict__ kf,
    const _Float16* __restrict__ vf, const float* __restrict__ acum,
    const float* __restrict__ norm_w, const float* __restrict__ order_coeff,
    float* __restrict__ attn_out) {
  __shared__ float    acum_s[L_SEQ];       // 8 KB: whole head cumsum
  __shared__ _Float16 Ks[32][DHD];         // [key][dh]   4 KB (TDM dest)
  __shared__ _Float16 Vraw[32][DHD];       // [key][dh]   4 KB (async dest)
  __shared__ _Float16 Vs[DHD][32];         // [dh][key]   4 KB (B^T for P*V)
  __shared__ _Float16 Ps[8][16][32];       // wave-private P tiles, 8 KB
  __shared__ float    Os[8][16][DHD];      // O for RMSNorm epilogue, 32 KB

  const int h    = blockIdx.y;
  const int tile = blockIdx.x;
  const int tid  = threadIdx.x, wave = tid >> 5, lane = tid & 31;
  const int row0 = tile * 128;

  for (int i = tid; i < L_SEQ; i += 256) acum_s[i] = acum[h * L_SEQ + i];

  // Q fragments (16x64 per wave -> two 16x32 A-fragments) from global
  const int qrow = row0 + wave * 16 + (lane & 15);
  const _Float16* qg = qf + ((size_t)h * L_SEQ + qrow) * DHD;
  const int kb8 = (lane & 16) ? 8 : 0;
  v16h aq[2];
#pragma unroll
  for (int ks = 0; ks < 2; ++ks) {
    const _Float16* qp = qg + ks * 32;
#pragma unroll
    for (int e = 0; e < 8; ++e) aq[ks][e] = qp[kb8 + e];
#pragma unroll
    for (int e = 0; e < 8; ++e) aq[ks][8 + e] = qp[16 + kb8 + e];
  }
  const float inv_oc = 1.0f / order_coeff[h];

  v8f vz = {};
  v8f acc[4];
#pragma unroll
  for (int i = 0; i < 4; ++i) acc[i] = vz;

  const int rb   = (lane & 16) ? 8 : 0;
  const int cofs = lane & 15;
  const int nkt  = (tile + 1) * 4;        // causal: key tiles up to row0+127

  // staging coordinates: one b128 per thread covers the 32x64 f16 tile
  const int i0   = tid * 8;
  const int keyA = i0 >> 6, dhA = i0 & 63;
  // transpose coordinates: read Vraw[keyT][dhT..dhT+7], scatter into Vs
  const int keyT = tid >> 3, dhT = (tid & 7) * 8;

  // incremental global addresses (advance by one 32x64 f16 tile per step)
  unsigned long long kga =
      (unsigned long long)(kf + ((size_t)h * L_SEQ) * DHD);       // TDM base
  const _Float16* vpt = vf + ((size_t)h * L_SEQ + keyA) * DHD + dhA;
  const unsigned ks_lds = (unsigned)(unsigned long long)&Ks[0][0];

  for (int j = 0; j < nkt; ++j) {
    __syncthreads();
    // --- stage: TDM for K, async b128 for V (raw) ---
    if (wave == 0) tdm_load_tile_32x64_f16(ks_lds, kga);
    async_copy_b128(vpt, &Vraw[keyA][dhA]);
    if (j + 1 < nkt) {                     // cache prefetch of next V tile
      __builtin_prefetch(vpt + 32 * DHD, 0, 1);
    }
    wait_async0();
    if (wave == 0) __builtin_amdgcn_s_wait_tensorcnt(0);
    __syncthreads();

    // --- LDS->LDS transpose of V while S is computed ---
#pragma unroll
    for (int e = 0; e < 8; ++e) Vs[dhT + e][keyT] = Vraw[keyT][dhT + e];

    // S = Q * K^T : 16x32 per wave = 2 N-tiles, K-depth 64 = 2 steps
    v8f s[2]; s[0] = vz; s[1] = vz;
#pragma unroll
    for (int nt = 0; nt < 2; ++nt)
#pragma unroll
      for (int ks = 0; ks < 2; ++ks) {
        v16h bf = frag_b_lds(&Ks[nt * 16][ks * 32], DHD);
        s[nt] = __builtin_amdgcn_wmma_f32_16x16x32_f16(
            false, aq[ks], false, bf, (short)0, s[nt], false, false);
      }

    // gate: causal * exp(Acum[row]-Acum[col]) * 1/order_coeff -> P (f16)
    const int kb = j * 32;
#pragma unroll
    for (int nt = 0; nt < 2; ++nt) {
      int colg = kb + nt * 16 + cofs;
      float ac_col = acum_s[colg];
#pragma unroll
      for (int r = 0; r < 8; ++r) {
        int rowl = rb + r;
        int rowg = row0 + wave * 16 + rowl;
        float m = (colg <= rowg) ? inv_oc : 0.f;   // cndmask, no branch
        float p = s[nt][r] * __expf(acum_s[rowg] - ac_col) * m;
        Ps[wave][rowl][nt * 16 + cofs] = (_Float16)p;
      }
    }
    __syncthreads();   // Vs transpose complete across all waves

    // O += P * V : A-frag from wave-private LDS, 4 B-frags over DH
    v16h ap = frag_a_lds(&Ps[wave][0][0], 32);
#pragma unroll
    for (int nt = 0; nt < 4; ++nt) {
      v16h bv = frag_b_lds(&Vs[nt * 16][0], 32);
      acc[nt] = __builtin_amdgcn_wmma_f32_16x16x32_f16(
          false, ap, false, bv, (short)0, acc[nt], false, false);
    }

    kga += (unsigned long long)(32 * DHD * 2);
    vpt += 32 * DHD;
  }

  // RMSNorm epilogue over DH, write attn_out[L][H*DH] (f32)
#pragma unroll
  for (int nt = 0; nt < 4; ++nt)
#pragma unroll
    for (int r = 0; r < 8; ++r)
      Os[wave][rb + r][nt * 16 + cofs] = acc[nt][r];
  __syncthreads();

  if (tid < 128) {
    int rowg = row0 + tid;
    const float* orow = &Os[tid >> 4][tid & 15][0];
    float ss = 0.f;
#pragma unroll
    for (int d = 0; d < DHD; ++d) { float x = orow[d]; ss += x * x; }
    float sc = rsqrtf(ss * (1.0f / DHD) + EPSF);
    float* og = attn_out + (size_t)rowg * HIDN + h * DHD;
#pragma unroll
    for (int d = 0; d < DHD; ++d) og[d] = orow[d] * sc * norm_w[d];
  }
}

// ---------------------------------------------------------------------------
extern "C" void kernel_launch(void* const* d_in, const int* in_sizes, int n_in,
                              void* d_out, int out_size, void* d_ws, size_t ws_size,
                              hipStream_t stream) {
  const float* hidden     = (const float*)d_in[0];
  // d_in[1] = attention_mask (causal, recomputed analytically)
  const float* qkv_w      = (const float*)d_in[2];
  const float* q_conv_w   = (const float*)d_in[3];
  const float* q_conv_b   = (const float*)d_in[4];
  const float* k_conv_w   = (const float*)d_in[5];
  const float* k_conv_b   = (const float*)d_in[6];
  const float* v_conv_w   = (const float*)d_in[7];
  const float* v_conv_b   = (const float*)d_in[8];
  const float* norm_w     = (const float*)d_in[9];
  const float* A_log      = (const float*)d_in[10];
  const float* dt_bias    = (const float*)d_in[11];
  const float* dt_proj_w  = (const float*)d_in[12];
  const float* dt_proj_b  = (const float*)d_in[13];
  const float* o_w        = (const float*)d_in[14];
  const float* order_coef = (const float*)d_in[15];

  char* ws = (char*)d_ws;
  size_t off = 0;
  float*    qkv  = (float*)(ws + off);    off += (size_t)L_SEQ * 3 * QS * 4;  // 24 MB
  float*    dt   = (float*)(ws + off);    off += (size_t)L_SEQ * NH * 4;
  float*    acum = (float*)(ws + off);    off += (size_t)NH * L_SEQ * 4;
  _Float16* qf   = (_Float16*)(ws + off); off += (size_t)L_SEQ * QS * 2;
  _Float16* kf   = (_Float16*)(ws + off); off += (size_t)L_SEQ * QS * 2;
  _Float16* vf   = (_Float16*)(ws + off); off += (size_t)L_SEQ * QS * 2;
  float*    attn = (float*)(ws + off);    off += (size_t)L_SEQ * HIDN * 4;

  // 1. QKV projection: (2048x1024) @ (3072x1024)^T
  gemm_f16wmma<3 * QS, HIDN><<<dim3(3 * QS / 128, L_SEQ / 128), 256, 0, stream>>>(
      hidden, qkv_w, qkv);
  // 2. dt gate
  dt_kernel<<<(L_SEQ * NH) / 256, 256, 0, stream>>>(
      hidden, dt_proj_w, dt_proj_b, dt_bias, dt);
  // 3. per-head cumulative decay
  cumsum_kernel<<<1, NH, 0, stream>>>(dt, A_log, acum);
  // 4. causal depthwise conv + head split + v*dt, f32->f16
  conv_split_kernel<<<(L_SEQ * QS) / 256, 256, 0, stream>>>(
      qkv, q_conv_w, q_conv_b, k_conv_w, k_conv_b, v_conv_w, v_conv_b, dt,
      qf, kf, vf);
  // 5. gated attention + RMSNorm
  attn_kernel<<<dim3(L_SEQ / 128, NH), 256, 0, stream>>>(
      qf, kf, vf, acum, norm_w, order_coef, attn);
  // 6. output projection: (2048x1024) @ (1024x1024)^T -> d_out
  gemm_f16wmma<HIDN, HIDN><<<dim3(HIDN / 128, L_SEQ / 128), 256, 0, stream>>>(
      attn, o_w, (float*)d_out);
}
